// WeightOnlyInt4Linear_55585466745295
// MI455X (gfx1250) — matrix operand order repaired and checked by reading, not compiled, not run on hardware
//
#include <hip/hip_runtime.h>
#include <stdint.h>

// WMMA operand vector types (gfx1250, wave32)
typedef __attribute__((ext_vector_type(16))) _Float16 v16h;
typedef __attribute__((ext_vector_type(8)))  float    v8f;
typedef __attribute__((ext_vector_type(4)))  _Float16 h4;

#define M_TOT 8192    // B*S
#define K_DIM 4096
#define N_DIM 11008
#define GSZ   128     // quant group size

#define BM 128
#define BN 256
#define BK 32
#define NSTEP (K_DIM / BK)   // 128 K-steps
#define LDA32 36      // fp32 LDS row stride in dwords
#define LDH   40      // f16 LDS row stride in halves (80 B = 16*5: b128-aligned rows)

// CDNA5 async copy: global -> LDS, tracked by ASYNCcnt
__device__ __forceinline__ void async_b128_to_lds(uint32_t lds_off, const float* g) {
    asm volatile("global_load_async_to_lds_b128 %0, %1, off"
                 :: "v"(lds_off), "v"(g) : "memory");
}
// in-order ASYNCcnt: <=4 -> previous 4-deep batch landed, next batch stays in flight
__device__ __forceinline__ void wait_async_le4() {
    asm volatile("s_wait_asynccnt 0x4" ::: "memory");
}
__device__ __forceinline__ void wait_async_le0() {
    asm volatile("s_wait_asynccnt 0x0" ::: "memory");
}

__global__ __launch_bounds__(256)
void wo_int4_gemm_wmma(const float* __restrict__ x,
                       const int*   __restrict__ wq,
                       const float* __restrict__ snz,
                       float*       __restrict__ out)
{
    __shared__ float    As32[2][BM * LDA32]; // raw fp32 X, async-DMA double buffer
    __shared__ _Float16 Ah[2][BM * LDH];     // f16 X tiles, double buffer
    __shared__ _Float16 Bh[2][BN * LDH];     // f16 dequantized W tiles, double buffer

    const int tid  = threadIdx.x;
    const int lane = tid & 31;
    const int wid  = tid >> 5;
    const int wm   = wid & 1;    // wave row (2)  -> 64 M-rows each
    const int wn   = wid >> 1;   // wave col (4)  -> 64 N-cols each

    const int tileM = blockIdx.y * BM;
    const int tileN = blockIdx.x * BN;

    // staging coordinates: 8 threads x 4 elems per 32-wide row
    const int lr = tid >> 3;          // 0..31 (+i*32)
    const int lc = (tid & 7) << 2;    // 0,4,...,28

    const uint32_t a32_base = (uint32_t)(uintptr_t)(&As32[0][0]);
    const uint32_t a32_span = (uint32_t)(BM * LDA32 * 4);

    v8f acc[4][4];
#pragma unroll
    for (int mi = 0; mi < 4; ++mi)
#pragma unroll
        for (int ni = 0; ni < 4; ++ni)
#pragma unroll
            for (int e = 0; e < 8; ++e)
                acc[mi][ni][e] = 0.0f;

    // per-lane fragment coordinates (CDNA5 ISA 7.12.2 layouts)
    const int amrow = wm * 64 + (lane & 15);
    const int kh    = (lane >> 4) * 8;
    const int bncol = wn * 64 + (lane & 15);
    const int kb    = (lane >> 4) * 8;

    union Frag { v16h v; h4 h[4]; };

    // issue async DMA for X tile of K-step `ks` into As32[slot]
    auto issueX = [&](int slot, int ks) {
#pragma unroll
        for (int i = 0; i < 4; ++i) {
            const int r = lr + i * 32;
            async_b128_to_lds(a32_base + (uint32_t)slot * a32_span
                                       + (uint32_t)((r * LDA32 + lc) * 4),
                              x + (size_t)(tileM + r) * K_DIM + ks * BK + lc);
        }
    };
    // convert own async-staged elements fp32->f16 (no cross-wave dependency)
    auto cvtX = [&](int slot, int p) {
#pragma unroll
        for (int i = 0; i < 4; ++i) {
            const int r = lr + i * 32;
            const float4 f = *(const float4*)(&As32[slot][r * LDA32 + lc]);
            h4 hv;
            hv.x = (_Float16)f.x; hv.y = (_Float16)f.y;
            hv.z = (_Float16)f.z; hv.w = (_Float16)f.w;
            *(h4*)(&Ah[p][r * LDH + lc]) = hv;
        }
    };
    // int4 codes -> fused groupwise dequant -> f16 LDS
    auto stageW = [&](int ks, int p) {
        const int g  = ks >> 2;       // 4 K-steps per quant group
        const int k0 = ks * BK;
#pragma unroll
        for (int j = 0; j < 8; ++j) {
            const int nn = lr + j * 32;
            const float2 sz = *(const float2*)(snz + ((size_t)g * N_DIM + tileN + nn) * 2);
            const float s = sz.x;
            const float z = sz.y - 8.0f * sz.x;   // fold (q-8)*s+z -> q*s+z'
            const int4 q = *(const int4*)(wq + (size_t)(tileN + nn) * K_DIM + k0 + lc);
            h4 hv;
            hv.x = (_Float16)fmaf((float)q.x, s, z);
            hv.y = (_Float16)fmaf((float)q.y, s, z);
            hv.z = (_Float16)fmaf((float)q.z, s, z);
            hv.w = (_Float16)fmaf((float)q.w, s, z);
            *(h4*)(&Bh[p][nn * LDH + lc]) = hv;
        }
    };
    // fragments + 16 WMMAs from f16 buffers p
    auto compute = [&](int p) {
        Frag a[4], b[4];
#pragma unroll
        for (int mi = 0; mi < 4; ++mi) {
            const _Float16* q = &Ah[p][(amrow + mi * 16) * LDH];
            a[mi].h[0] = *(const h4*)(q + kh);
            a[mi].h[1] = *(const h4*)(q + kh + 4);
            a[mi].h[2] = *(const h4*)(q + kh + 16);
            a[mi].h[3] = *(const h4*)(q + kh + 20);
        }
#pragma unroll
        for (int ni = 0; ni < 4; ++ni) {
            const _Float16* q = &Bh[p][(bncol + ni * 16) * LDH + kb];
            b[ni].h[0] = *(const h4*)(q);
            b[ni].h[1] = *(const h4*)(q + 4);
            b[ni].h[2] = *(const h4*)(q + 16);
            b[ni].h[3] = *(const h4*)(q + 20);
        }
#pragma unroll
        for (int mi = 0; mi < 4; ++mi)
#pragma unroll
            for (int ni = 0; ni < 4; ++ni)
                acc[mi][ni] = __builtin_amdgcn_wmma_f32_16x16x32_f16(
                    false, a[mi].v, false, b[ni].v,
                    (short)0, acc[mi][ni], false, false);
    };

    // ---- prologue: prime step 0, launch step 1 ----
    issueX(0, 0);
    wait_async_le0();
    cvtX(0, 0);
    stageW(0, 0);
    issueX(1, 1);         // X_1 in flight across the barrier
    __syncthreads();      // publish Ah[0]/Bh[0]

    // ---- main pipeline: one barrier per K-step ----
    int p = 0;
#pragma unroll 1
    for (int t = 0; t < NSTEP - 1; ++t) {
        const int t2 = (t + 2 < NSTEP) ? (t + 2) : (NSTEP - 1);  // branchless tail clamp
        issueX(t & 1, t2);       // X_{t+2} -> As32[t&1] (X_t there is fully consumed)
        wait_async_le4();        // X_{t+1} has landed in As32[(t+1)&1]
        cvtX((t + 1) & 1, p ^ 1);
        stageW(t + 1, p ^ 1);
        compute(p);              // 16 WMMAs overlap the staging above
        __syncthreads();         // publish p^1; retire frag reads of p
        p ^= 1;
    }
    compute(p);                  // peeled final K-step (dangling async retired by ENDPGM)

    // ---- epilogue: C/D layout -> global fp32 ----
    const int orow = tileM + wm * 64 + (lane >> 4) * 8;
    const int ocol = tileN + wn * 64 + (lane & 15);
#pragma unroll
    for (int mi = 0; mi < 4; ++mi)
#pragma unroll
        for (int ni = 0; ni < 4; ++ni)
#pragma unroll
            for (int e = 0; e < 8; ++e)
                out[(size_t)(orow + mi * 16 + e) * N_DIM + (ocol + ni * 16)] = acc[mi][ni][e];
}

extern "C" void kernel_launch(void* const* d_in, const int* in_sizes, int n_in,
                              void* d_out, int out_size, void* d_ws, size_t ws_size,
                              hipStream_t stream) {
    (void)in_sizes; (void)n_in; (void)d_ws; (void)ws_size; (void)out_size;
    const float* x   = (const float*)d_in[0];
    const int*   wq  = (const int*)d_in[1];
    const float* snz = (const float*)d_in[2];
    float* out = (float*)d_out;

    dim3 grid(N_DIM / BN, M_TOT / BM);   // 43 x 64 CTAs
    dim3 block(256);                     // 8 wave32
    hipLaunchKernelGGL(wo_int4_gemm_wmma, grid, block, 0, stream, x, wq, snz, out);
}
